// MultiHeadAttention_42502996361438
// MI455X (gfx1250) — compile-verified, matrix operand order checked by reading, not run on hardware
//
#include <hip/hip_runtime.h>
#include <hip/hip_bf16.h>

// MI455X / gfx1250, wave32. All matrix math via v_wmma_f32_16x16x32_f16.

typedef __attribute__((ext_vector_type(16))) _Float16 v16h;
typedef __attribute__((ext_vector_type(8)))  _Float16 v8h;
typedef __attribute__((ext_vector_type(4)))  _Float16 v4h;
typedef __attribute__((ext_vector_type(8)))  float    v8f;
typedef __attribute__((ext_vector_type(4)))  float    v4f;

#define B_DIM 2
#define C_DIM 2048
#define M_DIM 1024
#define H_DIM 16
#define K_DIM 64
#define V_DIM 64

__device__ __forceinline__ v8f wmma32(v16h a, v16h b, v8f c) {
  // D = A(16x32 f16) * B(32x16 f16) + C(16x16 f32)
  return __builtin_amdgcn_wmma_f32_16x16x32_f16(false, a, false, b, (short)0, c,
                                                false, false);
}

// Fragment gather for 16-bit WMMA operands.
// Works for A from row-major [M][K] storage and for B from row-major [N][K]
// storage (lane&15 = row, per-lane K chunks at 8*half and 16+8*half).
// Two b128 loads per lane; rows must be 16B aligned (ld % 8 == 0).
__device__ __forceinline__ v16h frag_ld_h(const _Float16* src, int ld, int lane) {
  const int r = lane & 15;
  const int half = lane >> 4;
  const _Float16* p = src + (size_t)r * ld + half * 8;
  v8h lo = *(const v8h*)(p);
  v8h hi = *(const v8h*)(p + 16);
  v16h f;
#pragma unroll
  for (int i = 0; i < 8; ++i) { f[i] = lo[i]; f[i + 8] = hi[i]; }
  return f;
}

// D-tile (16x16 f32 acc) stores. C/D layout: VGPR j, lane l -> m = j + 8*(l>>4),
// n = l & 15.
__device__ __forceinline__ void store_tile_h(_Float16* dst, int ld, int lane, v8f acc) {
  const int n = lane & 15, half = lane >> 4;
#pragma unroll
  for (int j = 0; j < 8; ++j)
    dst[(size_t)(j + 8 * half) * ld + n] = (_Float16)acc[j];
}

// Transposed store: dst indexed [n][m], m contiguous -> one b128 store per lane.
__device__ __forceinline__ void store_tile_hT(_Float16* dst, int ldT, int lane, v8f acc) {
  const int n = lane & 15, half = lane >> 4;
  v8h o;
#pragma unroll
  for (int j = 0; j < 8; ++j) o[j] = (_Float16)acc[j];
  *(v8h*)(dst + (size_t)n * ldT + 8 * half) = o;
}

__device__ __forceinline__ void store_tile_f(float* dst, int ld, int lane, v8f acc) {
  const int n = lane & 15, half = lane >> 4;
#pragma unroll
  for (int j = 0; j < 8; ++j)
    dst[(size_t)(j + 8 * half) * ld + n] = acc[j];
}

// ---------------------------------------------------------------------------
// Elementwise f32 -> f16 (vectorized x4).
__global__ __launch_bounds__(256)
void convert_f32_to_f16(const float* __restrict__ src, _Float16* __restrict__ dst,
                        size_t n4) {
  size_t i = ((size_t)blockIdx.x * 256 + threadIdx.x);
  if (i >= n4) return;
  v4f v = *(const v4f*)(src + i * 4);
  v4h o;
#pragma unroll
  for (int j = 0; j < 4; ++j) o[j] = (_Float16)v[j];
  *(v4h*)(dst + i * 4) = o;
}

// dst[batch][c][r] = (f16) src[batch][r][c]
__global__ __launch_bounds__(256)
void transpose_f32_to_f16(const float* __restrict__ src, _Float16* __restrict__ dst,
                          int R, int Ccols, size_t sBatch, size_t dBatch) {
  size_t idx = (size_t)blockIdx.x * blockDim.x + threadIdx.x;
  size_t total = (size_t)R * Ccols;
  if (idx >= total) return;
  int r = (int)(idx / Ccols);
  int c = (int)(idx % Ccols);
  const float* s = src + (size_t)blockIdx.z * sBatch;
  _Float16* d = dst + (size_t)blockIdx.z * dBatch;
  d[(size_t)c * R + r] = (_Float16)s[idx];
}

// ---------------------------------------------------------------------------
// y[b,h,c,k] = sum_m x16[b,c,m] * wT[h,k,m]
// Block tile: 128 c-rows x 64 k-cols, 8 waves, each wave 32x32 (4 WMMA tiles).
// transpose_out==0: y row-major [b][h][c][64]; ==1: y transposed [b][h][64][C]
__global__ __launch_bounds__(256)
void proj_kernel(const _Float16* __restrict__ x16, const _Float16* __restrict__ wT,
                 _Float16* __restrict__ y, int transpose_out) {
  const int b = blockIdx.z, h = blockIdx.y;
  const int wave = threadIdx.x >> 5;
  const int lane = threadIdx.x & 31;
  const int cw = blockIdx.x * 128 + (wave & 3) * 32;  // 32 output rows
  const int n0 = (wave >> 2) * 32;                    // 32 output cols

  const _Float16* xb = x16 + ((size_t)b * C_DIM + cw) * M_DIM;
  const _Float16* wh = wT + ((size_t)h * K_DIM + n0) * M_DIM;

  v8f acc00 = {}, acc01 = {}, acc10 = {}, acc11 = {};
#pragma unroll 2
  for (int m0 = 0; m0 < M_DIM; m0 += 32) {
    v16h a0 = frag_ld_h(xb + m0, M_DIM, lane);
    v16h a1 = frag_ld_h(xb + (size_t)16 * M_DIM + m0, M_DIM, lane);
    v16h b0 = frag_ld_h(wh + m0, M_DIM, lane);
    v16h b1 = frag_ld_h(wh + (size_t)16 * M_DIM + m0, M_DIM, lane);
    acc00 = wmma32(a0, b0, acc00);
    acc01 = wmma32(a0, b1, acc01);
    acc10 = wmma32(a1, b0, acc10);
    acc11 = wmma32(a1, b1, acc11);
  }

  const size_t bh = (size_t)(b * H_DIM + h);
  if (!transpose_out) {
    _Float16* yb = y + (bh * C_DIM + cw) * K_DIM;
    store_tile_h(yb + n0, K_DIM, lane, acc00);
    store_tile_h(yb + n0 + 16, K_DIM, lane, acc01);
    store_tile_h(yb + (size_t)16 * K_DIM + n0, K_DIM, lane, acc10);
    store_tile_h(yb + (size_t)16 * K_DIM + n0 + 16, K_DIM, lane, acc11);
  } else {
    _Float16* yb = y + (bh * V_DIM + n0) * C_DIM + cw;
    store_tile_hT(yb, C_DIM, lane, acc00);
    store_tile_hT(yb + (size_t)16 * C_DIM, C_DIM, lane, acc01);
    store_tile_hT(yb + 16, C_DIM, lane, acc10);
    store_tile_hT(yb + (size_t)16 * C_DIM + 16, C_DIM, lane, acc11);
  }
}

// ---------------------------------------------------------------------------
// Flash-style attention per (b,h). Block = 4 waves, each wave owns 16 queries.
// S[d,c] = q_d . k_c ; mask (c>d ? 0 : -100) added, then * 1/sqrt(64);
// online softmax over c (all keys, faithful to the -100 mask); acc += P * V.
__global__ __launch_bounds__(128)
void attn_kernel(const _Float16* __restrict__ qv,  // [B][H][C][64]
                 const _Float16* __restrict__ kv,  // [B][H][C][64]
                 const _Float16* __restrict__ vT,  // [B][H][64][C]
                 _Float16* __restrict__ pre) {     // [B][C][H][64]
  __shared__ __align__(16) float    sS[4][16][32];
  __shared__ __align__(16) _Float16 sP[4][16][32];
  __shared__ float sM[4][16];
  __shared__ float sL[4][16];
  __shared__ float sCorr[4][16];

  const int b = blockIdx.z, h = blockIdx.y;
  const int wave = threadIdx.x >> 5;
  const int lane = threadIdx.x & 31;
  const int n = lane & 15, half = lane >> 4;
  const int d0 = blockIdx.x * 64 + wave * 16;
  const size_t bh = (size_t)(b * H_DIM + h);

  const _Float16* qbase = qv + (bh * C_DIM + d0) * K_DIM;
  const _Float16* kbase = kv + bh * C_DIM * K_DIM;
  const _Float16* vbase = vT + bh * V_DIM * C_DIM;

  // Q fragments (k = 0..31 and 32..63), reused across all key tiles.
  const v16h aq0 = frag_ld_h(qbase, K_DIM, lane);
  const v16h aq1 = frag_ld_h(qbase + 32, K_DIM, lane);

  v8f acc[4] = {v8f{}, v8f{}, v8f{}, v8f{}};
  if (lane < 16) { sM[wave][lane] = -1e30f; sL[wave][lane] = 0.0f; }
  __syncthreads();

  for (int c0 = 0; c0 < C_DIM; c0 += 32) {
    // ---- S tiles: 16 queries x 32 keys (2 tiles x 2 k-steps) ----
    v8f s0 = {}, s1 = {};
    {
      v16h bk;
      bk = frag_ld_h(kbase + (size_t)c0 * K_DIM, K_DIM, lane);
      s0 = wmma32(aq0, bk, s0);
      bk = frag_ld_h(kbase + (size_t)c0 * K_DIM + 32, K_DIM, lane);
      s0 = wmma32(aq1, bk, s0);
      bk = frag_ld_h(kbase + (size_t)(c0 + 16) * K_DIM, K_DIM, lane);
      s1 = wmma32(aq0, bk, s1);
      bk = frag_ld_h(kbase + (size_t)(c0 + 16) * K_DIM + 32, K_DIM, lane);
      s1 = wmma32(aq1, bk, s1);
    }
    // ---- mask + scale, spill to LDS ----
#pragma unroll
    for (int j = 0; j < 8; ++j) {
      const int dl = j + 8 * half;
      const int dg = d0 + dl;
      const float m0 = (c0 + n > dg) ? 0.0f : -100.0f;
      const float m1 = (c0 + 16 + n > dg) ? 0.0f : -100.0f;
      sS[wave][dl][n]      = (s0[j] + m0) * 0.125f;
      sS[wave][dl][16 + n] = (s1[j] + m1) * 0.125f;
    }
    __syncthreads();
    // ---- online softmax row update (one lane per query row) ----
    if (lane < 16) {
      const float mOld = sM[wave][lane];
      float mNew = mOld;
#pragma unroll
      for (int c = 0; c < 32; ++c) mNew = fmaxf(mNew, sS[wave][lane][c]);
      const float corr = __expf(mOld - mNew);
      float lsum = 0.0f;
#pragma unroll
      for (int c = 0; c < 32; ++c) {
        const float p = __expf(sS[wave][lane][c] - mNew);
        sP[wave][lane][c] = (_Float16)p;
        lsum += p;
      }
      sM[wave][lane] = mNew;
      sL[wave][lane] = sL[wave][lane] * corr + lsum;
      sCorr[wave][lane] = corr;
    }
    __syncthreads();
    // ---- rescale accumulators and accumulate P * V ----
    float cf[8];
#pragma unroll
    for (int j = 0; j < 8; ++j) cf[j] = sCorr[wave][j + 8 * half];
#pragma unroll
    for (int t = 0; t < 4; ++t)
#pragma unroll
      for (int j = 0; j < 8; ++j) acc[t][j] *= cf[j];

    const v16h ap = frag_ld_h(&sP[wave][0][0], 32, lane);  // A: 16d x 32c
#pragma unroll
    for (int t = 0; t < 4; ++t) {
      v16h bv = frag_ld_h(vbase + (size_t)(16 * t) * C_DIM + c0, C_DIM, lane);
      acc[t] = wmma32(ap, bv, acc[t]);
    }
  }

  // ---- normalize and store pre[b, d, h, v] ----
  float linv[8];
#pragma unroll
  for (int j = 0; j < 8; ++j) linv[j] = 1.0f / sL[wave][j + 8 * half];

  _Float16* pbase =
      pre + ((size_t)b * C_DIM + d0) * (H_DIM * V_DIM) + (size_t)h * V_DIM;
#pragma unroll
  for (int t = 0; t < 4; ++t)
#pragma unroll
    for (int j = 0; j < 8; ++j) {
      const int dl = j + 8 * half;
      pbase[(size_t)dl * (H_DIM * V_DIM) + t * 16 + n] =
          (_Float16)(acc[t][j] * linv[j]);
    }
}

// ---------------------------------------------------------------------------
// out[bc, m] = sum_hv pre[bc, hv] * woT[m, hv]   (f32 output)
// Block tile: 128 rows x 64 cols, 8 waves, each wave 32x32 (4 WMMA tiles).
__global__ __launch_bounds__(256)
void out_kernel(const _Float16* __restrict__ pre,  // [B*C][H*V]
                const _Float16* __restrict__ woT,  // [M][H*V]
                float* __restrict__ out) {         // [B*C][M]
  const int wave = threadIdx.x >> 5;
  const int lane = threadIdx.x & 31;
  const int rw = blockIdx.x * 128 + (wave & 3) * 32;
  const int n0 = blockIdx.y * 64 + (wave >> 2) * 32;
  const int KR = H_DIM * V_DIM;  // 1024

  const _Float16* abase = pre + (size_t)rw * KR;
  const _Float16* bbase = woT + (size_t)n0 * KR;

  v8f acc00 = {}, acc01 = {}, acc10 = {}, acc11 = {};
#pragma unroll 2
  for (int k0 = 0; k0 < KR; k0 += 32) {
    v16h a0 = frag_ld_h(abase + k0, KR, lane);
    v16h a1 = frag_ld_h(abase + (size_t)16 * KR + k0, KR, lane);
    v16h b0 = frag_ld_h(bbase + k0, KR, lane);
    v16h b1 = frag_ld_h(bbase + (size_t)16 * KR + k0, KR, lane);
    acc00 = wmma32(a0, b0, acc00);
    acc01 = wmma32(a0, b1, acc01);
    acc10 = wmma32(a1, b0, acc10);
    acc11 = wmma32(a1, b1, acc11);
  }
  float* ob = out + (size_t)rw * M_DIM + n0;
  store_tile_f(ob, M_DIM, lane, acc00);
  store_tile_f(ob + 16, M_DIM, lane, acc01);
  store_tile_f(ob + (size_t)16 * M_DIM, M_DIM, lane, acc10);
  store_tile_f(ob + (size_t)16 * M_DIM + 16, M_DIM, lane, acc11);
}

// ---------------------------------------------------------------------------
extern "C" void kernel_launch(void* const* d_in, const int* in_sizes, int n_in,
                              void* d_out, int out_size, void* d_ws, size_t ws_size,
                              hipStream_t stream) {
  (void)in_sizes; (void)n_in; (void)out_size; (void)ws_size;
  const float* kvinput = (const float*)d_in[0];
  const float* qinput  = (const float*)d_in[1];
  const float* wq = (const float*)d_in[2];
  const float* wk = (const float*)d_in[3];
  const float* wv = (const float*)d_in[4];
  const float* wo = (const float*)d_in[5];
  float* out = (float*)d_out;

  char* ws = (char*)d_ws;
  size_t off = 0;
  auto alloc = [&](size_t bytes) -> char* {
    char* p = ws + off;
    off += (bytes + 255) & ~(size_t)255;
    return p;
  };
  const size_t WSZ = (size_t)H_DIM * K_DIM * M_DIM * sizeof(_Float16);          // 2 MB
  const size_t XSZ = (size_t)B_DIM * C_DIM * M_DIM * sizeof(_Float16);          // 8 MB
  const size_t QSZ = (size_t)B_DIM * H_DIM * C_DIM * K_DIM * sizeof(_Float16);  // 8 MB
  _Float16* wqT   = (_Float16*)alloc(WSZ);  // [H][64][M]
  _Float16* wkT   = (_Float16*)alloc(WSZ);  // [H][64][M]
  _Float16* wvT   = (_Float16*)alloc(WSZ);  // [H][64][M]
  _Float16* woT   = (_Float16*)alloc((size_t)M_DIM * H_DIM * V_DIM * sizeof(_Float16));  // [M][HV]
  _Float16* xq16  = (_Float16*)alloc(XSZ);  // [B][C][M] f16 qinput
  _Float16* xkv16 = (_Float16*)alloc(XSZ);  // [B][C][M] f16 kvinput
  _Float16* qval  = (_Float16*)alloc(QSZ);  // [B][H][C][64]
  _Float16* kval  = (_Float16*)alloc(QSZ);  // [B][H][C][64]
  _Float16* vvalT = (_Float16*)alloc(QSZ);  // [B][H][64][C]
  _Float16* pre2  = (_Float16*)alloc(QSZ);  // [B][C][H][64]

  const dim3 blk256(256);

  // Activation conversions f32 -> f16 (done once; heads reuse f16 data).
  {
    size_t n4 = (size_t)B_DIM * C_DIM * M_DIM / 4;  // 1,048,576
    dim3 g((unsigned)((n4 + 255) / 256));
    convert_f32_to_f16<<<g, blk256, 0, stream>>>(qinput, xq16, n4);
    convert_f32_to_f16<<<g, blk256, 0, stream>>>(kvinput, xkv16, n4);
  }
  // Weight transposes: [H][M][64] -> [H][64][M]
  {
    dim3 g((M_DIM * K_DIM) / 256, 1, H_DIM);
    size_t st = (size_t)M_DIM * K_DIM;
    transpose_f32_to_f16<<<g, blk256, 0, stream>>>(wq, wqT, M_DIM, K_DIM, st, st);
    transpose_f32_to_f16<<<g, blk256, 0, stream>>>(wk, wkT, M_DIM, K_DIM, st, st);
    transpose_f32_to_f16<<<g, blk256, 0, stream>>>(wv, wvT, M_DIM, K_DIM, st, st);
  }
  // wo: [H*V][M] -> [M][H*V]
  {
    dim3 g((H_DIM * V_DIM * M_DIM) / 256, 1, 1);
    transpose_f32_to_f16<<<g, blk256, 0, stream>>>(wo, woT, H_DIM * V_DIM, M_DIM, 0, 0);
  }

  // Projections
  {
    dim3 g(C_DIM / 128, H_DIM, B_DIM);
    proj_kernel<<<g, blk256, 0, stream>>>(xq16,  wqT, qval,  0);
    proj_kernel<<<g, blk256, 0, stream>>>(xkv16, wkT, kval,  0);
    proj_kernel<<<g, blk256, 0, stream>>>(xkv16, wvT, vvalT, 1);
  }

  // Attention (flash-style, faithful -100 mask over all keys)
  {
    dim3 g(C_DIM / 64, H_DIM, B_DIM);
    attn_kernel<<<g, dim3(128), 0, stream>>>(qval, kval, vvalT, pre2);
  }

  // Output projection
  {
    dim3 g((B_DIM * C_DIM) / 128, M_DIM / 64, 1);
    out_kernel<<<g, blk256, 0, stream>>>(pre2, woT, out);
  }
}
